// KPAttention_15144054685883
// MI455X (gfx1250) — matrix operand order repaired and checked by reading, not compiled, not run on hardware
//
#include <hip/hip_runtime.h>
#include <hip/hip_bf16.h>
#include <stdint.h>

// ---------------- types ----------------
typedef __attribute__((ext_vector_type(8)))  float   v8f;
typedef __attribute__((ext_vector_type(16))) __bf16  v16bf;
typedef __attribute__((ext_vector_type(8)))  __bf16  v8bf;

#define NB   17      // tokens
#define DIM  512
#define NH   8
#define HD   64
#define BATCH 4096
#define EPSV 1e-5f

// ws float offsets
#define WS_PART_SUM 0        // [64][512]
#define WS_PART_SQ  32768    // [64][512]
#define WS_SCALE    65536    // [512]
#define WS_SHIFT    66048    // [512]
#define WS_WQ_BYTES 266240   // bf16 qkv fragments: 96*16*32*16 = 786432 elems
#define WS_WP_BYTES (266240 + 1572864) // bf16 proj fragments: 32*16*32*16 = 262144 elems

// A-fragment loader: 16x32 bf16 A in this file's contiguous k order (matches the B swizzle).
// lane = half*16 + m(0..15); elems 0..15 -> K = half*16 + e
__device__ __forceinline__ v16bf load_a_frag(const __bf16* p) {
    v8bf lo = *(const v8bf*)p;
    v8bf hi = *(const v8bf*)(p + 8);
    return __builtin_shufflevector(lo, hi, 0,1,2,3,4,5,6,7,8,9,10,11,12,13,14,15);
}

// ---------------- kernel 1: BN partial stats (recompute graph conv on the fly) ----------------
__global__ __launch_bounds__(256) void bn_stats_kernel(
    const float* __restrict__ x, const float* __restrict__ adj, const float* __restrict__ adj2,
    const float* __restrict__ W, const float* __restrict__ M, const float* __restrict__ gbias,
    float* __restrict__ ws)
{
    __shared__ float sAs[NB*NB];
    __shared__ float sx[64][NB*2];
    int t = threadIdx.x;
    for (int i = t; i < NB*NB; i += 256) {
        int n = i / NB, m = i % NB;
        sAs[i] = 0.5f*(adj[n*NB+m]+adj2[n*NB+m]+adj[m*NB+n]+adj2[m*NB+n]);
    }
    size_t b0 = (size_t)blockIdx.x * 64;
    for (int i = t; i < 64*NB*2; i += 256)
        sx[i/(NB*2)][i%(NB*2)] = x[b0*NB*2 + i];
    __syncthreads();

    for (int cc = 0; cc < 2; ++cc) {
        int c = t + cc*256;
        float w00=W[c], w01=W[DIM+c], w10=W[2*DIM+c], w11=W[3*DIM+c];
        float gb = gbias[c];
        float Mc[NB];
        #pragma unroll
        for (int n=0;n<NB;++n) Mc[n] = M[n*DIM+c];
        float s1 = 0.f, s2 = 0.f;
        for (int bl = 0; bl < 64; ++bl) {
            float gm[NB];
            #pragma unroll
            for (int m=0;m<NB;++m)
                gm[m] = Mc[m]*(sx[bl][2*m]*w10 + sx[bl][2*m+1]*w11);
            #pragma unroll
            for (int n=0;n<NB;++n) {
                float dAn = sAs[n*NB+n];
                float acc = dAn*Mc[n]*(sx[bl][2*n]*w00 + sx[bl][2*n+1]*w01) + gb - dAn*gm[n];
                #pragma unroll
                for (int m=0;m<NB;++m) acc += sAs[n*NB+m]*gm[m];
                s1 += acc; s2 += acc*acc;
            }
        }
        ws[WS_PART_SUM + blockIdx.x*DIM + c] = s1;
        ws[WS_PART_SQ  + blockIdx.x*DIM + c] = s2;
    }
}

// ---------------- kernel 2: finalize BN scale/shift ----------------
__global__ __launch_bounds__(256) void bn_finalize_kernel(
    const float* __restrict__ bn_g, const float* __restrict__ bn_b, float* __restrict__ ws)
{
    int c = blockIdx.x*256 + threadIdx.x;
    float s1 = 0.f, s2 = 0.f;
    for (int p = 0; p < 64; ++p) {
        s1 += ws[WS_PART_SUM + p*DIM + c];
        s2 += ws[WS_PART_SQ  + p*DIM + c];
    }
    const float inv = 1.0f / ((float)BATCH * (float)NB);
    float mean = s1*inv;
    float var  = s2*inv - mean*mean;
    float sc   = bn_g[c] * rsqrtf(var + EPSV);
    ws[WS_SCALE + c] = sc;
    ws[WS_SHIFT + c] = bn_b[c] - mean*sc;
}

// ---------------- kernel 3: pre-swizzle weights into WMMA B-fragment order (bf16) ----------------
// layout: frag[((ct*16 + kt)*32 + lane)*16 + e] = w[d][kk], d = ct*16+(lane&15), kk = kt*32+(lane>>4)*16+e
__global__ __launch_bounds__(256) void swizzle_kernel(
    const float* __restrict__ qkv_w, const float* __restrict__ proj_w,
    __bf16* __restrict__ wq, __bf16* __restrict__ wp)
{
    int idx = blockIdx.x*256 + threadIdx.x;   // 4096*256 = 1,048,576 = 786432 + 262144
    if (idx < 786432) {
        int e = idx & 15, lane = (idx>>4)&31, kt = (idx>>9)&15, ct = idx>>13;
        int d  = ct*16 + (lane & 15);
        int kk = kt*32 + (lane >> 4)*16 + e;
        wq[idx] = (__bf16)qkv_w[d*DIM + kk];
    } else {
        int j = idx - 786432;
        int e = j & 15, lane = (j>>4)&31, kt = (j>>9)&15, ct = j>>13;
        int d  = ct*16 + (lane & 15);
        int kk = kt*32 + (lane >> 4)*16 + e;
        wp[j] = (__bf16)proj_w[d*DIM + kk];
    }
}

// ---------------- kernel 4: fused graphconv+BN+relu+pos+LN+attention+proj+residual ----------------
__global__ __launch_bounds__(256) void fused_kernel(
    const float* __restrict__ x, const float* __restrict__ adj, const float* __restrict__ adj2,
    const float* __restrict__ W, const float* __restrict__ M, const float* __restrict__ gbias,
    const float* __restrict__ pos, const float* __restrict__ ln_g, const float* __restrict__ ln_b,
    const float* __restrict__ proj_b, const float* __restrict__ wsF,
    const __bf16* __restrict__ wq, const __bf16* __restrict__ wp,
    float* __restrict__ out)
{
    __shared__ alignas(32) float  xpf[NB*520];        // residual (xp), fp32, stride 520
    __shared__ alignas(32) __bf16 xn [32*DIM];        // LN output; later reused as attn output "o"
    __shared__ alignas(32) __bf16 qs [32*DIM];
    __shared__ alignas(32) __bf16 ks [32*DIM];
    __shared__ alignas(32) __bf16 vT [DIM*32];        // v transposed: [d][token]
    __shared__ alignas(32) __bf16 att[NH*32*32];      // per-head attn, zero padded to 32x32
    __shared__ float sc[NH*NB*20];                    // raw scores, stride 20
    __shared__ float sAs[NB*NB];
    __shared__ float sx[NB*2];

    const int t   = threadIdx.x;
    const int wv  = t >> 5;          // wave id = head id
    const int ln  = t & 31;          // lane
    const int half = ln >> 4;
    const int l15  = ln & 15;
    const size_t b = blockIdx.x;

    // ---- stage 0: adjacency, x, zero xn pad rows 17..31 ----
    for (int i = t; i < NB*NB; i += 256) {
        int n = i / NB, m = i % NB;
        sAs[i] = 0.5f*(adj[n*NB+m]+adj2[n*NB+m]+adj[m*NB+n]+adj2[m*NB+n]);
    }
    if (t < NB*2) sx[t] = x[b*NB*2 + t];
    {
        uint32_t* xz = (uint32_t*)(xn + NB*DIM);
        for (int i = t; i < (32-NB)*DIM/2; i += 256) xz[i] = 0u;
    }
    __syncthreads();

    // ---- stage 1: graph conv -> BN -> relu -> +pos  => xpf (residual) ----
    for (int cc = 0; cc < 2; ++cc) {
        int c = t + cc*256;
        float w00=W[c], w01=W[DIM+c], w10=W[2*DIM+c], w11=W[3*DIM+c];
        float gb = gbias[c], scA = wsF[WS_SCALE+c], shB = wsF[WS_SHIFT+c];
        float Mc[NB], gm[NB];
        #pragma unroll
        for (int m=0;m<NB;++m) {
            Mc[m] = M[m*DIM+c];
            gm[m] = Mc[m]*(sx[2*m]*w10 + sx[2*m+1]*w11);
        }
        #pragma unroll
        for (int n=0;n<NB;++n) {
            float dAn = sAs[n*NB+n];
            float acc = dAn*Mc[n]*(sx[2*n]*w00 + sx[2*n+1]*w01) + gb - dAn*gm[n];
            #pragma unroll
            for (int m=0;m<NB;++m) acc += sAs[n*NB+m]*gm[m];
            float xb = acc*scA + shB;
            float xr = xb > 0.f ? xb : 0.f;
            xpf[n*520 + c] = xr + pos[n*DIM + c];
        }
    }
    __syncthreads();

    // ---- stage 2: LayerNorm -> xn (bf16, rows 17..31 already zero) ----
    for (int r = wv; r < NB; r += 8) {
        float s1 = 0.f, s2 = 0.f;
        #pragma unroll
        for (int k2 = 0; k2 < 16; ++k2) {
            float v = xpf[r*520 + ln + k2*32];
            s1 += v; s2 += v*v;
        }
        #pragma unroll
        for (int m = 16; m > 0; m >>= 1) {
            s1 += __shfl_xor(s1, m, 32);
            s2 += __shfl_xor(s2, m, 32);
        }
        float mu   = s1 * (1.0f/512.0f);
        float var  = s2 * (1.0f/512.0f) - mu*mu;
        float rstd = rsqrtf(var + EPSV);
        #pragma unroll
        for (int k2 = 0; k2 < 16; ++k2) {
            int c = ln + k2*32;
            xn[r*DIM + c] = (__bf16)((xpf[r*520 + c] - mu)*rstd*ln_g[c] + ln_b[c]);
        }
    }
    __syncthreads();

    // ---- stage 3: QKV GEMM  [32x512] x [512x1536] via bf16 WMMA ----
    for (int T = wv; T < 192; T += 8) {      // 96 col-tiles x 2 row-tiles
        int ct = T >> 1, rt = T & 1;
        v8f acc = {};
        const __bf16* aRow = xn + (rt*16 + l15)*DIM + half*16;
        #pragma unroll
        for (int kt = 0; kt < 16; ++kt) {
            v16bf a  = load_a_frag(aRow + kt*32);
            v16bf bb = *(const v16bf*)(wq + (size_t)(ct*512 + kt*32 + ln)*16);
            acc = __builtin_amdgcn_wmma_f32_16x16x32_bf16(false, a, false, bb, (short)0, acc, false, false);
        }
        int d = ct*16 + l15;
        int rbase = rt*16 + half*8;
        if (d < DIM) {
            #pragma unroll
            for (int rr=0;rr<8;++rr) qs[(rbase+rr)*DIM + d] = (__bf16)acc[rr];
        } else if (d < 2*DIM) {
            int dd = d - DIM;
            #pragma unroll
            for (int rr=0;rr<8;++rr) ks[(rbase+rr)*DIM + dd] = (__bf16)acc[rr];
        } else {
            int dd = d - 2*DIM;
            #pragma unroll
            for (int rr=0;rr<8;++rr) vT[dd*32 + (rbase+rr)] = (__bf16)acc[rr];
        }
    }
    __syncthreads();

    // ---- stage 4a: per-head scores = q k^T * scale (wave wv owns head wv) ----
    {
        int h = wv;
        uint32_t* az = (uint32_t*)(att + h*1024);
        for (int i = ln; i < 512; i += 32) az[i] = 0u;   // zero padded attn slab
        #pragma unroll
        for (int nt = 0; nt < 2; ++nt) {
            #pragma unroll
            for (int rt = 0; rt < 2; ++rt) {
                v8f acc = {};
                #pragma unroll
                for (int kt = 0; kt < 2; ++kt) {
                    v16bf a  = load_a_frag(qs + (rt*16 + l15)*DIM + h*HD + kt*32 + half*16);
                    v16bf bb = *(const v16bf*)(ks + (nt*16 + l15)*DIM + h*HD + kt*32 + half*16);
                    acc = __builtin_amdgcn_wmma_f32_16x16x32_bf16(false, a, false, bb, (short)0, acc, false, false);
                }
                int col = nt*16 + l15;
                if (col < NB) {
                    #pragma unroll
                    for (int rr=0;rr<8;++rr) {
                        int row = rt*16 + half*8 + rr;
                        if (row < NB) sc[h*NB*20 + row*20 + col] = acc[rr]*0.125f;  // HD^-0.5
                    }
                }
            }
        }
    }
    __syncthreads();

    // ---- stage 4b: softmax (lane r handles row r) ----
    {
        int h = wv;
        if (ln < NB) {
            float mx = -1e30f;
            #pragma unroll
            for (int c2=0;c2<NB;++c2) mx = fmaxf(mx, sc[h*NB*20 + ln*20 + c2]);
            float ev[NB]; float ssum = 0.f;
            #pragma unroll
            for (int c2=0;c2<NB;++c2) { ev[c2] = __expf(sc[h*NB*20 + ln*20 + c2] - mx); ssum += ev[c2]; }
            float inv = 1.0f/ssum;
            #pragma unroll
            for (int c2=0;c2<NB;++c2) att[h*1024 + ln*32 + c2] = (__bf16)(ev[c2]*inv);
        }
    }
    __syncthreads();

    // ---- stage 4c: o = attn @ v  (K padded to 32; vT cols >=17 are zero) ----
    {
        int h = wv;
        #pragma unroll
        for (int rt = 0; rt < 2; ++rt) {
            #pragma unroll
            for (int ct = 0; ct < 4; ++ct) {
                v8f acc = {};
                v16bf a  = load_a_frag(att + h*1024 + (rt*16 + l15)*32 + half*16);
                v16bf bb = *(const v16bf*)(vT + (h*HD + ct*16 + l15)*32 + half*16);
                acc = __builtin_amdgcn_wmma_f32_16x16x32_bf16(false, a, false, bb, (short)0, acc, false, false);
                int dcol = h*HD + ct*16 + l15;
                #pragma unroll
                for (int rr=0;rr<8;++rr) {
                    int row = rt*16 + half*8 + rr;
                    if (row < NB) xn[row*DIM + dcol] = (__bf16)acc[rr];  // reuse xn as o-buffer
                }
            }
        }
    }
    __syncthreads();

    // ---- stage 5: proj GEMM + bias + residual -> d_out ----
    for (int T = wv; T < 64; T += 8) {       // 32 col-tiles x 2 row-tiles
        int ct = T >> 1, rt = T & 1;
        v8f acc = {};
        const __bf16* aRow = xn + (rt*16 + l15)*DIM + half*16;
        #pragma unroll
        for (int kt = 0; kt < 16; ++kt) {
            v16bf a  = load_a_frag(aRow + kt*32);
            v16bf bb = *(const v16bf*)(wp + (size_t)(ct*512 + kt*32 + ln)*16);
            acc = __builtin_amdgcn_wmma_f32_16x16x32_bf16(false, a, false, bb, (short)0, acc, false, false);
        }
        int d  = ct*16 + l15;
        float pb = proj_b[d];
        #pragma unroll
        for (int rr=0;rr<8;++rr) {
            int row = rt*16 + half*8 + rr;
            if (row < NB)
                out[b*NB*DIM + row*DIM + d] = acc[rr] + pb + xpf[row*520 + d];
        }
    }
}

// ---------------- launcher ----------------
extern "C" void kernel_launch(void* const* d_in, const int* in_sizes, int n_in,
                              void* d_out, int out_size, void* d_ws, size_t ws_size,
                              hipStream_t stream) {
    const float* x      = (const float*)d_in[0];
    const float* adj    = (const float*)d_in[1];
    const float* adj2   = (const float*)d_in[2];
    const float* W      = (const float*)d_in[3];
    const float* M      = (const float*)d_in[4];
    const float* gbias  = (const float*)d_in[5];
    const float* bn_g   = (const float*)d_in[6];
    const float* bn_b   = (const float*)d_in[7];
    const float* pos    = (const float*)d_in[8];
    const float* ln_g   = (const float*)d_in[9];
    const float* ln_b   = (const float*)d_in[10];
    const float* qkv_w  = (const float*)d_in[11];
    const float* proj_w = (const float*)d_in[12];
    const float* proj_b = (const float*)d_in[13];

    float*  wsF = (float*)d_ws;
    __bf16* wq  = (__bf16*)((char*)d_ws + WS_WQ_BYTES);
    __bf16* wp  = (__bf16*)((char*)d_ws + WS_WP_BYTES);
    float*  out = (float*)d_out;

    bn_stats_kernel   <<<64,   256, 0, stream>>>(x, adj, adj2, W, M, gbias, wsF);
    bn_finalize_kernel<<<2,    256, 0, stream>>>(bn_g, bn_b, wsF);
    swizzle_kernel    <<<4096, 256, 0, stream>>>(qkv_w, proj_w, wq, wp);
    fused_kernel      <<<4096, 256, 0, stream>>>(x, adj, adj2, W, M, gbias, pos,
                                                 ln_g, ln_b, proj_b, wsF, wq, wp, out);
}
